// CausalSelfAttention_90168543412762
// MI455X (gfx1250) — matrix operand order repaired and checked by reading, not compiled, not run on hardware
//
#include <hip/hip_runtime.h>
#include <hip/hip_bf16.h>

typedef __attribute__((ext_vector_type(16))) _Float16 v16h;
typedef __attribute__((ext_vector_type(2)))  _Float16 v2h;
typedef __attribute__((ext_vector_type(8)))  float    v8f;

#define N_EMBD 1024
#define N_HEAD 16
#define HD     64
#define BATCH  4
#define SEQT   2048
#define MROWS  (BATCH * SEQT)   // 8192

__device__ __forceinline__ v8f wmma16x16x32(v16h a, v16h b, v8f c) {
  // (neg_a, A, neg_b, B, c_mod, C, reuse_a, reuse_b)
  return __builtin_amdgcn_wmma_f32_16x16x32_f16(false, a, false, b, (short)0, c,
                                                false, false);
}

// -------------------------------------------------------------------------
// fp32 -> f16 conversion
// -------------------------------------------------------------------------
__global__ void cvt_f32_f16(const float* __restrict__ src,
                            _Float16* __restrict__ dst, int n) {
  int i = blockIdx.x * blockDim.x + threadIdx.x;
  if (i < n) dst[i] = (_Float16)src[i];
}

// -------------------------------------------------------------------------
// GEMM: out[m,n] = sum_k A[m,k] * Bw[n,k]   (Bw is torch-Linear [out,in])
// 64x64 tile per 128-thread block; 4 waves (2x2), each wave 32x32 (2x2 wmma).
// -------------------------------------------------------------------------
template <bool OUT_F32>
__global__ __launch_bounds__(128)
void gemm_f16(const _Float16* __restrict__ A, const _Float16* __restrict__ Bw,
              float* __restrict__ Cf, _Float16* __restrict__ Ch,
              const float* __restrict__ bias, int M, int N, int K) {
  __shared__ __align__(16) _Float16 As[64 * 32];
  __shared__ __align__(16) _Float16 Bs[64 * 32];

  const int tid  = threadIdx.x;
  const int lane = tid & 31;
  const int w    = tid >> 5;
  const int wr   = w >> 1;
  const int wc   = w & 1;
  const int m0   = blockIdx.y * 64;
  const int n0   = blockIdx.x * 64;

  v8f acc[2][2] = {};

  const int nkt = K >> 5;
  for (int kt = 0; kt < nkt; ++kt) {
    __syncthreads();
    // cooperative tile load: 64 rows x 32 halves each, as 16B chunks
#pragma unroll
    for (int i = 0; i < 2; ++i) {
      int u   = tid + i * 128;     // 0..255
      int row = u >> 2;            // 0..63
      int seg = (u & 3) << 3;      // {0,8,16,24} halves
      *(uint4*)(&As[row * 32 + seg]) =
          *(const uint4*)(A + (size_t)(m0 + row) * K + (kt << 5) + seg);
      *(uint4*)(&Bs[row * 32 + seg]) =
          *(const uint4*)(Bw + (size_t)(n0 + row) * K + (kt << 5) + seg);
    }
    __syncthreads();

    v16h af[2], bf[2];
#pragma unroll
    for (int s = 0; s < 2; ++s) {
      // A fragment (16x32, M x K): lane<16 -> K 0..15, lane>=16 -> K 16..31
      int arow = wr * 32 + s * 16 + (lane & 15);
      int akb  = ((lane >> 4) & 1) * 16;
      const _Float16* ab = &As[arow * 32];
#pragma unroll
      for (int p = 0; p < 8; ++p) {
        int k   = akb + (p << 1);
        v2h t   = *(const v2h*)(ab + k);
        af[s][2 * p]     = t.x;
        af[s][2 * p + 1] = t.y;
      }
      // B fragment (32x16, K x N): lane<16 -> K 0..15, lane>=16 -> K 16..31
      int brow = wc * 32 + s * 16 + (lane & 15);
      int bkb  = ((lane >> 4) & 1) * 16;
      const _Float16* bb = &Bs[brow * 32 + bkb];
#pragma unroll
      for (int p = 0; p < 8; ++p) {
        v2h t = *(const v2h*)(bb + (p << 1));
        bf[s][2 * p]     = t.x;
        bf[s][2 * p + 1] = t.y;
      }
    }

#pragma unroll
    for (int sm = 0; sm < 2; ++sm)
#pragma unroll
      for (int sn = 0; sn < 2; ++sn)
        acc[sm][sn] = wmma16x16x32(af[sm], bf[sn], acc[sm][sn]);
  }

  // epilogue: C/D layout -> (m = r + 8*(lane>>4), n = lane&15)
#pragma unroll
  for (int sm = 0; sm < 2; ++sm)
#pragma unroll
    for (int sn = 0; sn < 2; ++sn)
#pragma unroll
      for (int r = 0; r < 8; ++r) {
        int m   = m0 + wr * 32 + sm * 16 + r + ((lane >> 4) << 3);
        int n   = n0 + wc * 32 + sn * 16 + (lane & 15);
        float v = acc[sm][sn][r];
        if (OUT_F32)
          Cf[(size_t)m * N + n] = v + bias[n];
        else
          Ch[(size_t)m * N + n] = (_Float16)v;
      }
}

// -------------------------------------------------------------------------
// Flash-style causal attention. One block = (batch b, head h, 64 q-rows).
// 4 waves; wave w owns q rows [w*16, w*16+16). Online softmax, f16 WMMA.
// NOTE: reference multiplies scores by sqrt(hd) = 8 (faithful).
// -------------------------------------------------------------------------
__global__ __launch_bounds__(128)
void attn_kernel(const _Float16* __restrict__ qkv, _Float16* __restrict__ Y) {
  __shared__ __align__(16) _Float16 Qs[64 * 64];
  __shared__ __align__(16) _Float16 Ks[64 * 64];
  __shared__ __align__(16) _Float16 Vs[64 * 64];
  __shared__ __align__(16) _Float16 Ps[4 * 16 * 64];

  const int tid   = threadIdx.x;
  const int lane  = tid & 31;
  const int w     = tid >> 5;
  const int qt    = blockIdx.x;
  const int hh    = blockIdx.y;
  const int bz    = blockIdx.z;
  const int qbase = qt * 64;
  const size_t rowbase = (size_t)(bz * SEQT) * 3072;

  // stage Q tile (64 x 64 halves)
#pragma unroll
  for (int i = 0; i < 4; ++i) {
    int u   = tid + i * 128;     // 0..511
    int row = u >> 3;            // 0..63
    int seg = (u & 7) << 3;      // 0..56 halves
    *(uint4*)(&Qs[row * 64 + seg]) =
        *(const uint4*)(qkv + rowbase + (size_t)(qbase + row) * 3072 +
                        hh * 64 + seg);
  }
  __syncthreads();

  // persistent Q A-fragments (two K-chunks of 32 over hd=64)
  v16h qa[2];
  {
    int row = w * 16 + (lane & 15);
    int kb  = ((lane >> 4) & 1) * 16;
#pragma unroll
    for (int c = 0; c < 2; ++c) {
      const _Float16* base = &Qs[row * 64 + c * 32];
#pragma unroll
      for (int p = 0; p < 8; ++p) {
        int k = kb + (p << 1);
        v2h t = *(const v2h*)(base + k);
        qa[c][2 * p]     = t.x;
        qa[c][2 * p + 1] = t.y;
      }
    }
  }

  v8f of[4] = {};
  float m_run[8], l_run[8];
#pragma unroll
  for (int r = 0; r < 8; ++r) {
    m_run[r] = -__builtin_inff();
    l_run[r] = 0.0f;
  }

  const int nkt = qt + 1;  // causal: only key tiles with j0 <= qbase
  for (int jt = 0; jt < nkt; ++jt) {
    const int j0 = jt * 64;
    __syncthreads();
#pragma unroll
    for (int i = 0; i < 4; ++i) {
      int u   = tid + i * 128;
      int row = u >> 3;
      int seg = (u & 7) << 3;
      size_t g = rowbase + (size_t)(j0 + row) * 3072 + hh * 64 + seg;
      *(uint4*)(&Ks[row * 64 + seg]) = *(const uint4*)(qkv + g + 1024);
      *(uint4*)(&Vs[row * 64 + seg]) = *(const uint4*)(qkv + g + 2048);
    }
    __syncthreads();

    // S = Q * K^T  (scaled by 8), causal-masked
    float sv[4][8];
#pragma unroll
    for (int sn = 0; sn < 4; ++sn) {
      v8f sacc = {};
#pragma unroll
      for (int kc = 0; kc < 2; ++kc) {
        v16h bf;
        int key = sn * 16 + (lane & 15);
        int kb  = kc * 32 + ((lane >> 4) & 1) * 16;
        const _Float16* base = &Ks[key * 64 + kb];
#pragma unroll
        for (int p = 0; p < 8; ++p) {
          v2h t = *(const v2h*)(base + (p << 1));
          bf[2 * p]     = t.x;
          bf[2 * p + 1] = t.y;
        }
        sacc = wmma16x16x32(qa[kc], bf, sacc);
      }
#pragma unroll
      for (int r = 0; r < 8; ++r) {
        int qi  = qbase + w * 16 + r + ((lane >> 4) << 3);
        int kj  = j0 + sn * 16 + (lane & 15);
        float s = sacc[r] * 8.0f;  // * sqrt(hd), faithful to reference
        sv[sn][r] = (kj > qi) ? -__builtin_inff() : s;
      }
    }

    // online softmax per q-row (rows live across 16-lane groups)
#pragma unroll
    for (int r = 0; r < 8; ++r) {
      float rmax = fmaxf(fmaxf(sv[0][r], sv[1][r]), fmaxf(sv[2][r], sv[3][r]));
#pragma unroll
      for (int off = 8; off > 0; off >>= 1)
        rmax = fmaxf(rmax, __shfl_xor(rmax, off, 16));
      float mnew = fmaxf(m_run[r], rmax);
      float corr = __expf(m_run[r] - mnew);
      float psum = 0.0f;
#pragma unroll
      for (int sn = 0; sn < 4; ++sn) {
        float p   = __expf(sv[sn][r] - mnew);
        sv[sn][r] = p;
        psum += p;
      }
#pragma unroll
      for (int off = 8; off > 0; off >>= 1)
        psum += __shfl_xor(psum, off, 16);
      l_run[r] = l_run[r] * corr + psum;
      m_run[r] = mnew;
#pragma unroll
      for (int d = 0; d < 4; ++d) of[d][r] *= corr;
      int mw = r + ((lane >> 4) << 3);
#pragma unroll
      for (int sn = 0; sn < 4; ++sn)
        Ps[w * 1024 + mw * 64 + sn * 16 + (lane & 15)] = (_Float16)sv[sn][r];
    }
    __syncthreads();  // make P visible for frag re-layout

    // O += P * V
    v16h pa[2];
    {
      int row = lane & 15;
      int kb  = ((lane >> 4) & 1) * 16;
#pragma unroll
      for (int c = 0; c < 2; ++c) {
        const _Float16* base = &Ps[w * 1024 + row * 64 + c * 32];
#pragma unroll
        for (int p = 0; p < 8; ++p) {
          int k = kb + (p << 1);
          v2h t = *(const v2h*)(base + k);
          pa[c][2 * p]     = t.x;
          pa[c][2 * p + 1] = t.y;
        }
      }
    }
#pragma unroll
    for (int d = 0; d < 4; ++d) {
#pragma unroll
      for (int kc = 0; kc < 2; ++kc) {
        v16h vb;
        int dd = d * 16 + (lane & 15);
        int kb = kc * 32 + ((lane >> 4) & 1) * 16;
#pragma unroll
        for (int h = 0; h < 16; ++h) vb[h] = Vs[(kb + h) * 64 + dd];
        of[d] = wmma16x16x32(pa[kc], vb, of[d]);
      }
    }
  }

  // normalize and emit y (f16, [8192, 1024])
#pragma unroll
  for (int d = 0; d < 4; ++d)
#pragma unroll
    for (int r = 0; r < 8; ++r) {
      float v  = of[d][r] / l_run[r];
      int rowg = bz * SEQT + qbase + w * 16 + r + ((lane >> 4) << 3);
      int col  = hh * 64 + d * 16 + (lane & 15);
      Y[(size_t)rowg * 1024 + col] = (_Float16)v;
    }
}

// -------------------------------------------------------------------------
// launch
// -------------------------------------------------------------------------
extern "C" void kernel_launch(void* const* d_in, const int* in_sizes, int n_in,
                              void* d_out, int out_size, void* d_ws,
                              size_t ws_size, hipStream_t stream) {
  const float* x      = (const float*)d_in[0];
  const float* W_attn = (const float*)d_in[1];
  const float* W_proj = (const float*)d_in[2];
  const float* b_proj = (const float*)d_in[3];
  float* out          = (float*)d_out;

  auto alignup = [](size_t v) { return (v + 255) & ~(size_t)255; };
  char* p = (char*)d_ws;
  _Float16* x_h   = (_Float16*)p; p += alignup((size_t)MROWS * N_EMBD * 2);
  _Float16* wq_h  = (_Float16*)p; p += alignup((size_t)3 * N_EMBD * N_EMBD * 2);
  _Float16* wp_h  = (_Float16*)p; p += alignup((size_t)N_EMBD * N_EMBD * 2);
  _Float16* qkv_h = (_Float16*)p; p += alignup((size_t)MROWS * 3 * N_EMBD * 2);
  _Float16* y_h   = (_Float16*)p;

  const int n1 = MROWS * N_EMBD;        // x
  const int n2 = 3 * N_EMBD * N_EMBD;   // W_attn
  const int n3 = N_EMBD * N_EMBD;       // W_proj
  cvt_f32_f16<<<(n1 + 255) / 256, 256, 0, stream>>>(x, x_h, n1);
  cvt_f32_f16<<<(n2 + 255) / 256, 256, 0, stream>>>(W_attn, wq_h, n2);
  cvt_f32_f16<<<(n3 + 255) / 256, 256, 0, stream>>>(W_proj, wp_h, n3);

  // qkv = x @ W_attn^T   [8192, 3072] f16
  gemm_f16<false><<<dim3(3 * N_EMBD / 64, MROWS / 64), 128, 0, stream>>>(
      x_h, wq_h, nullptr, qkv_h, nullptr, MROWS, 3 * N_EMBD, N_EMBD);

  // y = softmax(causal(Q K^T * 8)) V    [8192, 1024] f16
  attn_kernel<<<dim3(SEQT / 64, N_HEAD, BATCH), 128, 0, stream>>>(qkv_h, y_h);

  // out = y @ W_proj^T + b    [8192, 1024] fp32
  gemm_f16<true><<<dim3(N_EMBD / 64, MROWS / 64), 128, 0, stream>>>(
      y_h, wp_h, out, nullptr, b_proj, MROWS, N_EMBD, N_EMBD);
}
